// RBGHead_23922967838986
// MI455X (gfx1250) — matrix operand order repaired and checked by reading, not compiled, not run on hardware
//
#include <hip/hip_runtime.h>
#include <math.h>

// ---------------- problem constants ----------------
#define BN     2
#define NPROP  64
#define NSEEDN 2048
#define NPTSN  4096
#define FPSM   1024
#define CDIMN  256
#define RCN    64
#define HCN    32
#define AGGN   128
#define RNUM   66
#define BINSN  6
#define NSN    8
#define BRN    (BINSN*RNUM)     /* 396  */
#define QTOT   (NPROP*BRN)      /* 25344 */
#define PI_F   3.14159265358979323846f

typedef __attribute__((ext_vector_type(16))) _Float16 v16h;
typedef __attribute__((ext_vector_type(8)))  _Float16 v8h;
typedef __attribute__((ext_vector_type(8)))  float    v8f;

// =====================================================================
// RAY direction table (matches generate_ray(66): n=4)
// =====================================================================
__global__ void rbg_ray_kernel(float* __restrict__ ray){
  if (threadIdx.x != 0 || blockIdx.x != 0) return;
  const int n = 4;
  float azim = 0.0f;
  const float astep = 0.5f * PI_F / (float)n;
  int o = 0;
  for (int a = -n; a <= n; ++a){
    int aa = (a < 0) ? -a : a;
    int size = (n - aa) * 4; if (size == 0) size = 1;
    float step = 2.0f * PI_F / (float)size;
    float pol = 0.0f;
    float sa = sinf(azim), ca = cosf(azim);
    for (int i = 0; i < size; ++i){
      pol += step;
      ray[o*3+0] = cosf(pol) * sa;
      ray[o*3+1] = sinf(pol) * sa;
      ray[o*3+2] = ca;
      ++o;
    }
    azim += astep;
  }
}

// =====================================================================
// Farthest point sampling: one block (256 thr) per batch, 16 pts/thread
// =====================================================================
__global__ void rbg_fps_kernel(const float* __restrict__ pts, int* __restrict__ fidx){
  const int b = blockIdx.x;
  const int t = threadIdx.x;                   // 256 threads
  const float* P = pts + (long long)b * NPTSN * 3;
  float px[16], py[16], pz[16], dl[16];
  #pragma unroll
  for (int j = 0; j < 16; ++j){
    int g = t*16 + j;
    px[j] = P[g*3+0]; py[j] = P[g*3+1]; pz[j] = P[g*3+2];
    dl[j] = 1e10f;
  }
  __shared__ float sval[256];
  __shared__ int   sidx[256];
  int last = 0;
  for (int it = 0; it < FPSM; ++it){
    if (t == 0) fidx[b*FPSM + it] = last;
    float cx = P[last*3+0], cy = P[last*3+1], cz = P[last*3+2];
    float bv = -1.0f; int bi = t*16;
    #pragma unroll
    for (int j = 0; j < 16; ++j){
      float dx = px[j]-cx, dy = py[j]-cy, dz = pz[j]-cz;
      float d = dx*dx + dy*dy + dz*dz;
      dl[j] = fminf(dl[j], d);
      if (dl[j] > bv){ bv = dl[j]; bi = t*16 + j; }   // first-index on ties
    }
    sval[t] = bv; sidx[t] = bi;
    __syncthreads();
    for (int s = 128; s > 0; s >>= 1){
      if (t < s){
        float ov = sval[t+s]; int oi = sidx[t+s];
        if (ov > sval[t] || (ov == sval[t] && oi < sidx[t])){ sval[t] = ov; sidx[t] = oi; }
      }
      __syncthreads();
    }
    last = sidx[0];
    __syncthreads();
  }
}

// gather tgt = points_cat[fidx]  (B*FPSM threads)
__global__ void rbg_gather_tgt(const float* __restrict__ pts, const int* __restrict__ fidx,
                               float* __restrict__ tgt){
  int i = blockIdx.x*blockDim.x + threadIdx.x;
  if (i >= BN*FPSM) return;
  int b = i / FPSM;
  int j = fidx[i];
  const float* p = pts + ((long long)b*NPTSN + j)*3;
  tgt[i*3+0] = p[0]; tgt[i*3+1] = p[1]; tgt[i*3+2] = p[2];
}

// =====================================================================
// 3-NN of each target among seeds (squared dists) + weights
// =====================================================================
__global__ void rbg_nn3_kernel(const float* __restrict__ tgt, const float* __restrict__ seed_xyz,
                               int* __restrict__ idx3, float* __restrict__ w3){
  int i = blockIdx.x*blockDim.x + threadIdx.x;
  if (i >= BN*FPSM) return;
  int b = i / FPSM;
  float tx = tgt[i*3+0], ty = tgt[i*3+1], tz = tgt[i*3+2];
  const float* S = seed_xyz + (long long)b * NSEEDN * 3;
  float v0 = 1e30f, v1 = 1e30f, v2 = 1e30f;
  int   i0 = 0,    i1 = 0,    i2 = 0;
  for (int s = 0; s < NSEEDN; ++s){
    float dx = S[s*3+0]-tx, dy = S[s*3+1]-ty, dz = S[s*3+2]-tz;
    float d = fmaxf(dx*dx + dy*dy + dz*dz, 0.0f);
    if (d < v0)      { v2=v1; i2=i1; v1=v0; i1=i0; v0=d; i0=s; }
    else if (d < v1) { v2=v1; i2=i1; v1=d;  i1=s; }
    else if (d < v2) { v2=d;  i2=s; }
  }
  float r0 = 1.0f/(v0+1e-8f), r1 = 1.0f/(v1+1e-8f), r2 = 1.0f/(v2+1e-8f);
  float inv = 1.0f/(r0+r1+r2);
  idx3[i*3+0]=i0; idx3[i*3+1]=i1; idx3[i*3+2]=i2;
  w3[i*3+0]=r0*inv; w3[i*3+1]=r1*inv; w3[i*3+2]=r2*inv;
}

// interp0[b][c][m] = sum_j w3 * seed_features[b][c][idx3]
__global__ void rbg_interp_kernel(const float* __restrict__ sf, const int* __restrict__ idx3,
                                  const float* __restrict__ w3, float* __restrict__ out){
  int i = blockIdx.x*blockDim.x + threadIdx.x;
  if (i >= BN*CDIMN*FPSM) return;
  int m = i % FPSM;
  int c = (i / FPSM) % CDIMN;
  int b = i / (CDIMN*FPSM);
  int base = (b*FPSM + m)*3;
  const float* row = sf + ((long long)b*CDIMN + c)*NSEEDN;
  out[i] = w3[base+0]*row[idx3[base+0]]
         + w3[base+1]*row[idx3[base+1]]
         + w3[base+2]*row[idx3[base+2]];
}

// =====================================================================
// GEMM staging: weights f32[M,K] -> f16[Mp,K] zero padded rows
// =====================================================================
__global__ void rbg_convW(const float* __restrict__ W, _Float16* __restrict__ Wh,
                          int M, int Mp, int K){
  int i = blockIdx.x*blockDim.x + threadIdx.x;
  if (i >= Mp*K) return;
  int m = i / K;
  Wh[i] = (m < M) ? (_Float16)W[i] : (_Float16)0.0f;
}

// activations f32[B][K,N] (batch stride sX) -> f16 col-major [B][N,K]
__global__ void rbg_convXT(const float* __restrict__ X, _Float16* __restrict__ XT,
                           int K, int N, long long sX){
  long long i = (long long)blockIdx.x*blockDim.x + threadIdx.x;
  if (i >= (long long)BN*N*K) return;
  int k = (int)(i % K);
  int n = (int)((i / K) % N);
  int b = (int)(i / ((long long)N*K));
  XT[i] = (_Float16)X[(long long)b*sX + (long long)k*N + n];
}

// =====================================================================
// WMMA GEMM (TN, f16 staged):  Y[b] = relu?( W[M,K] * X[b][K,N] + bias )
// 1 wave = one 16x16 tile.  Both fragments are K-contiguous per lane ->
// 4x global_load_b128 + 1x v_wmma_f32_16x16x32_f16 per K-step, no guards.
// =====================================================================
__global__ void rbg_gemm_wmma(const _Float16* __restrict__ Wh, const float* __restrict__ bias,
                              const _Float16* __restrict__ XT, float* __restrict__ Y,
                              int M, int K, int N, long long sXT, long long sY, int relu){
  int lane = threadIdx.x & 31;
  int wv   = threadIdx.x >> 5;
  int tn   = blockIdx.x * (blockDim.x >> 5) + wv;
  int tm   = blockIdx.y;
  int b    = blockIdx.z;
  if (tn * 16 >= N) return;
  const int l15 = lane & 15, kg = lane >> 4;
  // A fragment: row = tm*16+l15, halves [8*kg, 8*kg+8) and [8*kg+16, 8*kg+24)
  const _Float16* Ap = Wh + (long long)(tm*16 + l15) * K + 8*kg;
  // B fragment: col = tn*16+l15, halves [8*kg, 8*kg+8) and [8*kg+16, 8*kg+24)
  const _Float16* Bp = XT + (long long)b * sXT + (long long)(tn*16 + l15) * K + 8*kg;
  v8f acc = {};
  #pragma unroll 2
  for (int k0 = 0; k0 < K; k0 += 32){
    v8h a0 = *(const v8h*)(Ap);
    v8h a1 = *(const v8h*)(Ap + 16);
    v8h b0 = *(const v8h*)(Bp);
    v8h b1 = *(const v8h*)(Bp + 16);
    __builtin_prefetch((const void*)(Bp + 64), 0, 1);   // global_prefetch_b8
    v16h av = __builtin_shufflevector(a0, a1, 0,1,2,3,4,5,6,7,8,9,10,11,12,13,14,15);
    v16h bv = __builtin_shufflevector(b0, b1, 0,1,2,3,4,5,6,7,8,9,10,11,12,13,14,15);
    acc = __builtin_amdgcn_wmma_f32_16x16x32_f16(false, av, false, bv,
                                                 (short)0, acc, false, false);
    Ap += 32; Bp += 32;
  }
  // C/D epilogue: f32 16x16 layout (vgpr r -> M = r + 8*(lane>>4))
  int n = tn*16 + l15;
  float* Yb = Y + (long long)b * sY;
  #pragma unroll
  for (int r = 0; r < 8; ++r){
    int row = tm*16 + 8*kg + r;
    if (row < M && n < N){
      float y = acc[r] + bias[row];
      if (relu) y = fmaxf(y, 0.0f);
      Yb[(long long)row*N + n] = y;
    }
  }
}

// =====================================================================
// Query point generation
// =====================================================================
__global__ void rbg_qgen_coarse(const float* __restrict__ scale, const float* __restrict__ ref,
                                const float* __restrict__ ray, float* __restrict__ qxyz){
  int i = blockIdx.x*blockDim.x + threadIdx.x;          // (b,p,bin,r)
  if (i >= BN*QTOT) return;
  int r   = i % RNUM;
  int bin = (i / RNUM) % BINSN;
  int p   = (i / BRN) % NPROP;
  int b   = i / QTOT;
  float d = scale[b*NPROP + p] * (float)(bin+1) / (float)BINSN;
  const float* rp = ref + (b*NPROP + p)*3;
  qxyz[i*3+0] = rp[0] + d*ray[r*3+0];
  qxyz[i*3+1] = rp[1] + d*ray[r*3+1];
  qxyz[i*3+2] = rp[2] + d*ray[r*3+2];
}

__global__ void rbg_qgen_fine(const float* __restrict__ scale, const float* __restrict__ ref,
                              const float* __restrict__ ray, const int* __restrict__ cmask,
                              float* __restrict__ qxyz){
  int i = blockIdx.x*blockDim.x + threadIdx.x;
  if (i >= BN*QTOT) return;
  int r    = i % RNUM;
  int fbin = (i / RNUM) % BINSN;
  int p    = (i / BRN) % NPROP;
  int b    = i / QTOT;
  float s = scale[b*NPROP + p];
  float surf = 0.0f;                                    // max over bins of cdist*mask
  for (int bin = 0; bin < BINSN; ++bin)
    if (cmask[(long long)b*QTOT + p*BRN + bin*RNUM + r])
      surf = s * (float)(bin+1) / (float)BINSN;         // monotone in bin
  float d = surf * (float)(fbin+1) / (float)BINSN;
  const float* rp = ref + (b*NPROP + p)*3;
  qxyz[i*3+0] = rp[0] + d*ray[r*3+0];
  qxyz[i*3+1] = rp[1] + d*ray[r*3+1];
  qxyz[i*3+2] = rp[2] + d*ray[r*3+2];
}

// =====================================================================
// Fused SA module: ball-query (first-8-in-index-order) + 2-layer MLP +
// max-pool.  1 wave per query (wave32), 8 waves/block, weights in LDS.
// =====================================================================
__global__ __launch_bounds__(256)
void rbg_sagroup(const float* __restrict__ tgt, const float* __restrict__ interp,
                 const float* __restrict__ qxyz,
                 const float* __restrict__ W1, const float* __restrict__ B1,
                 const float* __restrict__ W2, const float* __restrict__ B2,
                 int* __restrict__ qidx, float* __restrict__ feat, float r2){
  __shared__ float sW1[RCN*(RCN+3)];   // 64 x 67
  __shared__ float sB1[RCN];
  __shared__ float sW2[HCN*RCN];       // 32 x 64
  __shared__ float sB2[HCN];
  __shared__ float sIn[8][68];
  __shared__ float sH1[8][RCN];

  const int tid  = threadIdx.x;
  const int wave = tid >> 5, lane = tid & 31;
  const int b    = blockIdx.y;
  const int q    = blockIdx.x*8 + wave;

  for (int i = tid; i < RCN*(RCN+3); i += 256) sW1[i] = W1[i];
  for (int i = tid; i < RCN;         i += 256) sB1[i] = B1[i];
  for (int i = tid; i < HCN*RCN;     i += 256) sW2[i] = W2[i];
  if (tid < HCN) sB2[tid] = B2[tid];
  __syncthreads();

  const float* T = tgt + (long long)b*FPSM*3;
  float qx = qxyz[((long long)b*QTOT + q)*3 + 0];
  float qy = qxyz[((long long)b*QTOT + q)*3 + 1];
  float qz = qxyz[((long long)b*QTOT + q)*3 + 2];

  // -------- ball query: first 8 in-radius targets in index order --------
  int idxs[NSN]; int cnt = 0; int first = -1;
  for (int base = 0; base < FPSM; base += 32){
    int j = base + lane;
    float dx = T[j*3+0]-qx, dy = T[j*3+1]-qy, dz = T[j*3+2]-qz;
    bool in = (dx*dx + dy*dy + dz*dz) < r2;
    unsigned m = __builtin_amdgcn_ballot_w32(in);
    while (m && cnt < NSN){
      int bp = __builtin_ctz(m); m &= m - 1u;
      int jj = base + bp;
      if (first < 0) first = jj;
      idxs[cnt++] = jj;
    }
    if (cnt >= NSN) break;
  }
  bool empty = (cnt == 0);
  if (first < 0) first = 0;
  for (int s = cnt; s < NSN; ++s) idxs[s] = first;
  if (lane < NSN) qidx[((long long)b*QTOT + q)*NSN + lane] = idxs[lane];

  // -------- grouped MLP (67 -> 64 relu -> 32 relu) + maxpool --------
  const float* IP = interp + (long long)b*RCN*FPSM;
  float fm = -1e30f;
  for (int s = 0; s < NSN; ++s){
    int j = idxs[s];
    __syncthreads();
    if (lane == 0) sIn[wave][0] = T[j*3+0] - qx;
    if (lane == 1) sIn[wave][1] = T[j*3+1] - qy;
    if (lane == 2) sIn[wave][2] = T[j*3+2] - qz;
    sIn[wave][3 + lane]      = IP[(lane     )*FPSM + j];
    sIn[wave][3 + lane + 32] = IP[(lane + 32)*FPSM + j];
    __syncthreads();
    float h1a = sB1[lane], h1b = sB1[lane + 32];
    #pragma unroll 1
    for (int c = 0; c < RCN+3; ++c){
      float x = sIn[wave][c];
      h1a += sW1[(lane     )*(RCN+3) + c] * x;
      h1b += sW1[(lane + 32)*(RCN+3) + c] * x;
    }
    sH1[wave][lane]      = fmaxf(h1a, 0.0f);
    sH1[wave][lane + 32] = fmaxf(h1b, 0.0f);
    __syncthreads();
    float h2 = sB2[lane];
    #pragma unroll 1
    for (int c = 0; c < RCN; ++c) h2 += sW2[lane*RCN + c] * sH1[wave][c];
    fm = fmaxf(fm, fmaxf(h2, 0.0f));
  }
  feat[((long long)b*HCN + lane)*QTOT + q] = empty ? 0.0f : fm;  // lane == channel
}

// concat(broadcast agg, feat) -> cat160 [B,160,Q]
__global__ void rbg_cat160(const float* __restrict__ aggf, const float* __restrict__ feat,
                           float* __restrict__ out){
  long long i = (long long)blockIdx.x*blockDim.x + threadIdx.x;
  if (i >= (long long)BN*(AGGN+HCN)*QTOT) return;
  int q = (int)(i % QTOT);
  int c = (int)((i / QTOT) % (AGGN+HCN));
  int b = (int)(i / ((long long)(AGGN+HCN)*QTOT));
  out[i] = (c < AGGN) ? aggf[((long long)b*AGGN + c)*NPROP + q/BRN]
                      : feat[((long long)b*HCN + (c-AGGN))*QTOT + q];
}

// argmax(scores) -> mask; write masked qidx (as float) to d_out
__global__ void rbg_msk(const float* __restrict__ sc, const int* __restrict__ qidx,
                        int* __restrict__ msk, float* __restrict__ outq){
  long long i = (long long)blockIdx.x*blockDim.x + threadIdx.x;
  if (i >= (long long)BN*QTOT) return;
  int q = (int)(i % QTOT);
  int b = (int)(i / QTOT);
  float s0 = sc[((long long)b*2 + 0)*QTOT + q];
  float s1 = sc[((long long)b*2 + 1)*QTOT + q];
  int m = (s1 > s0) ? 1 : 0;          // jnp.argmax ties -> 0
  msk[i] = m;
  #pragma unroll
  for (int s = 0; s < NSN; ++s)
    outq[i*NSN + s] = (float)(qidx[i*NSN + s] * m);
}

// masked rf rearranged: t192[b][h*6+bin][r*64+p] = feat[b][h][q]*msk[q]
__global__ void rbg_t192(const float* __restrict__ feat, const int* __restrict__ msk,
                         float* __restrict__ out){
  long long i = (long long)blockIdx.x*blockDim.x + threadIdx.x;
  const long long TOT = (long long)BN*(HCN*BINSN)*(RNUM*NPROP);
  if (i >= TOT) return;
  int col = (int)(i % (RNUM*NPROP));
  int row = (int)((i / (RNUM*NPROP)) % (HCN*BINSN));
  int b   = (int)(i / ((long long)(HCN*BINSN)*(RNUM*NPROP)));
  int h = row / BINSN, bin = row % BINSN;
  int r = col / NPROP, p   = col % NPROP;
  int q = p*BRN + bin*RNUM + r;
  out[i] = feat[((long long)b*HCN + h)*QTOT + q] * (float)msk[(long long)b*QTOT + q];
}

// concat(ffeat, cfeat) -> [B,256,NP]
__global__ void rbg_pack256(const float* __restrict__ ff, const float* __restrict__ cf,
                            float* __restrict__ out){
  int i = blockIdx.x*blockDim.x + threadIdx.x;
  if (i >= BN*CDIMN*NPROP) return;
  int p = i % NPROP;
  int c = (i / NPROP) % CDIMN;
  int b = i / (CDIMN*NPROP);
  out[i] = (c < AGGN) ? ff[(b*AGGN + c)*NPROP + p]
                      : cf[(b*AGGN + (c-AGGN))*NPROP + p];
}

// =====================================================================
// Host side
// =====================================================================
struct GemmStage { _Float16* Wh; _Float16* XT; };

static void launch_gemm(hipStream_t s, GemmStage st,
                        const float* W, const float* bias,
                        const float* X, float* Y, int M, int K, int N,
                        long long sX, long long sY, int relu){
  int Mp = (M + 15) & ~15;
  rbg_convW<<<(Mp*K + 255)/256, 256, 0, s>>>(W, st.Wh, M, Mp, K);
  long long nxt = (long long)BN*N*K;
  rbg_convXT<<<(unsigned)((nxt + 255)/256), 256, 0, s>>>(X, st.XT, K, N, sX);
  dim3 grd(((N/16) + 3)/4, Mp/16, BN);
  rbg_gemm_wmma<<<grd, 128, 0, s>>>(st.Wh, bias, st.XT, Y, M, K, N,
                                    (long long)N*K, sY, relu);
}

extern "C" void kernel_launch(void* const* d_in, const int* in_sizes, int n_in,
                              void* d_out, int out_size, void* d_ws, size_t ws_size,
                              hipStream_t stream){
  (void)in_sizes; (void)n_in; (void)out_size; (void)ws_size;
  const float* seed_xyz   = (const float*)d_in[0];
  const float* seed_feats = (const float*)d_in[1];
  const float* scale_pred = (const float*)d_in[3];
  const float* ref_points = (const float*)d_in[4];
  const float* points_cat = (const float*)d_in[5];
  const float* agg_feats  = (const float*)d_in[6];
  // params flattened in sorted-key pytree order, (W, b) per layer
  auto PR = [&](int i){ return (const float*)d_in[i]; };
  const float *cbinW = PR(7),  *cbinB = PR(8);
  const float *cinW0 = PR(9),  *cinB0 = PR(10), *cinW1 = PR(11), *cinB1 = PR(12);
  const float *crW0  = PR(13), *crB0  = PR(14), *crW1  = PR(15), *crB1  = PR(16);
  const float *csW0  = PR(17), *csB0  = PR(18), *csW1  = PR(19), *csB1  = PR(20);
  const float *fbinW = PR(21), *fbinB = PR(22);
  const float *finW0 = PR(23), *finB0 = PR(24), *finW1 = PR(25), *finB1 = PR(26);
  const float *frW0  = PR(27), *frB0  = PR(28), *frW1  = PR(29), *frB1  = PR(30);
  const float *fsW0  = PR(31), *fsB0  = PR(32), *fsW1  = PR(33), *fsB1  = PR(34);
  const float *fuW0  = PR(35), *fuB0  = PR(36), *fuW1  = PR(37), *fuB1  = PR(38);
  const float *srW0  = PR(39), *srB0  = PR(40), *srW1  = PR(41), *srB1  = PR(42);

  float* out = (float*)d_out;
  const long long OUT_FUSE = 0;
  const long long OUT_FSC  = (long long)BN*AGGN*NPROP;            // 16384
  const long long OUT_FQ   = OUT_FSC + (long long)BN*2*QTOT;      // +101376
  const long long OUT_CSC  = OUT_FQ  + (long long)BN*QTOT*NSN;    // +405504
  const long long OUT_CQ   = OUT_CSC + (long long)BN*2*QTOT;

  // ---- workspace carve-up (floats) ----
  float* ws = (float*)d_ws;
  long long cur = 0;
  auto alloc = [&](long long nf){ float* p = ws + cur; cur += (nf + 63) & ~(long long)63; return p; };
  float* ray     = alloc(RNUM*3);
  int*   fidx    = (int*)alloc(BN*FPSM);
  float* tgt     = alloc(BN*FPSM*3);
  int*   idx3    = (int*)alloc(BN*FPSM*3);
  float* w3      = alloc(BN*FPSM*3);
  float* interp0 = alloc((long long)BN*CDIMN*FPSM);
  float* tmp128  = alloc((long long)BN*128*FPSM);
  float* interp  = alloc((long long)BN*RCN*FPSM);
  float* qxyz    = alloc((long long)BN*QTOT*3);
  int*   qidx    = (int*)alloc((long long)BN*QTOT*NSN);
  float* feat    = alloc((long long)BN*HCN*QTOT);
  float* cat160  = alloc((long long)BN*(AGGN+HCN)*QTOT);
  float* h32     = alloc((long long)BN*HCN*QTOT);
  int*   msk     = (int*)alloc((long long)BN*QTOT);
  float* t192    = alloc((long long)BN*(HCN*BINSN)*(RNUM*NPROP));
  float* t32     = alloc((long long)BN*HCN*(RNUM*NPROP));
  float* ray1    = alloc((long long)BN*CDIMN*NPROP);
  float* cfeat   = alloc((long long)BN*AGGN*NPROP);
  float* ffeat   = alloc((long long)BN*AGGN*NPROP);
  float* fusecat = alloc((long long)BN*CDIMN*NPROP);
  float* fusemid = alloc((long long)BN*CDIMN*NPROP);
  // f16 staging (sized for the largest W [256x2112] and largest XT [B,Q,160])
  GemmStage st;
  st.Wh = (_Float16*)alloc((long long)(256*2112)/2 + 64);
  st.XT = (_Float16*)alloc(((long long)BN*QTOT*(AGGN+HCN))/2 + 64);

  // ---- front end: FPS + gather + 3NN interpolation + seed_reduce ----
  rbg_ray_kernel<<<1, 32, 0, stream>>>(ray);
  rbg_fps_kernel<<<BN, 256, 0, stream>>>(points_cat, fidx);
  rbg_gather_tgt<<<(BN*FPSM + 255)/256, 256, 0, stream>>>(points_cat, fidx, tgt);
  rbg_nn3_kernel<<<(BN*FPSM + 255)/256, 256, 0, stream>>>(tgt, seed_xyz, idx3, w3);
  rbg_interp_kernel<<<(BN*CDIMN*FPSM + 255)/256, 256, 0, stream>>>(seed_feats, idx3, w3, interp0);
  launch_gemm(stream, st, srW0, srB0, interp0, tmp128, 128, CDIMN, FPSM,
              (long long)CDIMN*FPSM, (long long)128*FPSM, 1);
  launch_gemm(stream, st, srW1, srB1, tmp128, interp, RCN, 128, FPSM,
              (long long)128*FPSM, (long long)RCN*FPSM, 0);

  // ---- branch runner (coarse then fine; scratch reused) ----
  auto run_branch = [&](bool fine,
                        const float* saW0, const float* saB0, const float* saW1, const float* saB1,
                        const float* inW0, const float* inB0, const float* inW1, const float* inB1,
                        const float* bnW,  const float* bnB,
                        const float* rW0,  const float* rB0,  const float* rW1,  const float* rB1,
                        float radius, float* scOut, float* qOut, float* bOut){
    int nq = BN*QTOT;
    if (!fine) rbg_qgen_coarse<<<(nq+255)/256, 256, 0, stream>>>(scale_pred, ref_points, ray, qxyz);
    else       rbg_qgen_fine  <<<(nq+255)/256, 256, 0, stream>>>(scale_pred, ref_points, ray, msk, qxyz);
    rbg_sagroup<<<dim3(QTOT/8, BN), 256, 0, stream>>>(tgt, interp, qxyz,
                                                      saW0, saB0, saW1, saB1,
                                                      qidx, feat, radius*radius);
    long long ncat = (long long)BN*(AGGN+HCN)*QTOT;
    rbg_cat160<<<(unsigned)((ncat + 255)/256), 256, 0, stream>>>(agg_feats, feat, cat160);
    launch_gemm(stream, st, inW0, inB0, cat160, h32, HCN, AGGN+HCN, QTOT,
                (long long)(AGGN+HCN)*QTOT, (long long)HCN*QTOT, 1);
    launch_gemm(stream, st, inW1, inB1, h32, scOut, 2, HCN, QTOT,
                (long long)HCN*QTOT, (long long)2*QTOT, 0);
    rbg_msk<<<(BN*QTOT + 255)/256, 256, 0, stream>>>(scOut, qidx, msk, qOut);
    long long nt = (long long)BN*(HCN*BINSN)*(RNUM*NPROP);
    rbg_t192<<<(unsigned)((nt + 255)/256), 256, 0, stream>>>(feat, msk, t192);
    launch_gemm(stream, st, bnW, bnB, t192, t32, HCN, HCN*BINSN, RNUM*NPROP,
                (long long)(HCN*BINSN)*(RNUM*NPROP), (long long)HCN*(RNUM*NPROP), 0);
    // t32 reinterpret [B, HC*R=2112, NP=64] is a free reshape (same memory)
    launch_gemm(stream, st, rW0, rB0, t32, ray1, CDIMN, HCN*RNUM, NPROP,
                (long long)(HCN*RNUM)*NPROP, (long long)CDIMN*NPROP, 1);
    launch_gemm(stream, st, rW1, rB1, ray1, bOut, AGGN, CDIMN, NPROP,
                (long long)CDIMN*NPROP, (long long)AGGN*NPROP, 0);
  };

  run_branch(false, csW0, csB0, csW1, csB1, cinW0, cinB0, cinW1, cinB1,
             cbinW, cbinB, crW0, crB0, crW1, crB1,
             0.8f, out + OUT_CSC, out + OUT_CQ, cfeat);
  // fine query-gen reads the coarse mask left in `msk` before overwriting it
  run_branch(true,  fsW0, fsB0, fsW1, fsB1, finW0, finB0, finW1, finB1,
             fbinW, fbinB, frW0, frB0, frW1, frB1,
             0.4f, out + OUT_FSC, out + OUT_FQ, ffeat);

  // ---- fuse ----
  rbg_pack256<<<(BN*CDIMN*NPROP + 255)/256, 256, 0, stream>>>(ffeat, cfeat, fusecat);
  launch_gemm(stream, st, fuW0, fuB0, fusecat, fusemid, CDIMN, CDIMN, NPROP,
              (long long)CDIMN*NPROP, (long long)CDIMN*NPROP, 1);
  launch_gemm(stream, st, fuW1, fuB1, fusemid, out + OUT_FUSE, AGGN, CDIMN, NPROP,
              (long long)CDIMN*NPROP, (long long)AGGN*NPROP, 0);
}
